// PNAAggregator_3341484556653
// MI455X (gfx1250) — matrix operand order repaired and checked by reading, not compiled, not run on hardware
//
#include <hip/hip_runtime.h>
#include <math.h>

typedef __attribute__((ext_vector_type(16))) _Float16 v16h;
typedef __attribute__((ext_vector_type(8)))  _Float16 v8h;
typedef __attribute__((ext_vector_type(8)))  float    v8f;

#define PNA_D      64
#define PNA_K      768          // 12 * D
#define PNA_OUT    64
#define NEG_SLOPE  0.2f
#define APITCH     776          // 768 + 8 halves pad -> breaks LDS bank conflicts
#define RED_BLOCKS 96

// ---------------------------------------------------------------- init
__global__ void pna_init(float* sum, float* sumsq, float* maxf, float* minf,
                         float* deg, float* misc, long md, int m) {
    long i = (long)blockIdx.x * blockDim.x + threadIdx.x;
    if (i < md) {
        sum[i]   = 0.f;
        sumsq[i] = 0.f;
        maxf[i]  = 0.f;                 // torch max starts from zeros
        minf[i]  = __builtin_inff();    // torch min starts from +inf
    }
    if (i < m)   deg[i]  = 0.f;
    if (i < 130) misc[i] = 0.f;         // reduction partials + delta slot
}

// ---------------------------------------------------------------- W -> f16
__global__ void pna_wconv(const float* __restrict__ W, _Float16* __restrict__ Wh, int n) {
    int i = blockIdx.x * blockDim.x + threadIdx.x;
    if (i < n) Wh[i] = (_Float16)W[i];
}

// ---------------------------------------------------------------- edge scatter
// one wave (32 lanes) per edge; each lane handles 2 features -> the 256B
// feature row is one coalesced load per edge.
__global__ void pna_edges(const int* __restrict__ rows, const int* __restrict__ cols,
                          const float* __restrict__ vals, const float* __restrict__ feat,
                          float* sum, float* sumsq, float* maxf, float* minf,
                          float* deg, int nnz) {
    long gid = (long)blockIdx.x * blockDim.x + threadIdx.x;
    int  e    = (int)(gid >> 5);
    int  lane = (int)(gid & 31);
    if (e >= nnz) return;
    int   r = rows[e];
    int   c = cols[e];
    float v = vals[e];
    float2 x = ((const float2*)(feat + (long)c * PNA_D))[lane];
    long base = (long)r * PNA_D + lane * 2;
    atomicAdd(sum + base,       v * x.x);
    atomicAdd(sum + base + 1,   v * x.y);
    atomicAdd(sumsq + base,     v * x.x * x.x);
    atomicAdd(sumsq + base + 1, v * x.y * x.y);
    atomicMax(maxf + base,      x.x);
    atomicMax(maxf + base + 1,  x.y);
    atomicMin(minf + base,      x.x);
    atomicMin(minf + base + 1,  x.y);
    if (lane == 0) atomicAdd(deg + r, v);
}

// ---------------------------------------------------------------- delta = mean(log10(deg+2))
__global__ void pna_logdeg_partial(const float* __restrict__ deg, float* partials, int m) {
    __shared__ float sh[256];
    float acc = 0.f;
    for (int i = blockIdx.x * blockDim.x + threadIdx.x; i < m; i += gridDim.x * blockDim.x)
        acc += log10f(deg[i] + 2.0f);
    sh[threadIdx.x] = acc;
    __syncthreads();
    for (int s = 128; s > 0; s >>= 1) {
        if (threadIdx.x < s) sh[threadIdx.x] += sh[threadIdx.x + s];
        __syncthreads();
    }
    if (threadIdx.x == 0) partials[blockIdx.x] = sh[0];
}

__global__ void pna_logdeg_final(const float* __restrict__ partials, float* delta, int nb, int m) {
    if (threadIdx.x == 0) {
        float t = 0.f;
        for (int i = 0; i < nb; ++i) t += partials[i];   // fixed order -> deterministic
        *delta = t / (float)m;
    }
}

// ---------------------------------------------------------------- fused feature-build + GEMM + bias + leaky
// One wave per 16-row tile. A-tile [16 x 768] built in LDS as f16, then
// out[16x64] = A @ Wh^T via 24 K-steps x 4 N-tiles of v_wmma_f32_16x16x32_f16.
__global__ __launch_bounds__(32)
void pna_gemm(const float* __restrict__ sum, const float* __restrict__ sumsq,
              const float* __restrict__ maxf, const float* __restrict__ minf,
              const float* __restrict__ deg, const float* __restrict__ deltaPtr,
              const _Float16* __restrict__ Wh, const float* __restrict__ bias,
              float* __restrict__ out, int m) {
    __shared__ alignas(16) _Float16 At[16 * APITCH];

    const int lane    = threadIdx.x;
    const int rowbase = blockIdx.x * 16;
    const float delta = *deltaPtr;

    // ---- build 16 x 768 f16 A-tile: [mean, mean*s, mean/s, max, ..., std/s]
    for (int idx = lane; idx < 16 * PNA_D; idx += 32) {
        int r = idx >> 6;
        int d = idx & 63;
        int gr = rowbase + r;
        if (gr >= m) gr = m - 1;                  // clamp (keeps EXEC full)
        float dg = deg[gr];
        float sd = dg > 0.f ? dg : 1.f;
        long  gb = (long)gr * PNA_D + d;
        float mean = sum[gb] / sd;
        float msq  = sumsq[gb] / sd;
        float var  = msq - mean * mean;
        if (var < 0.f) var = 0.f;
        float stdv = sqrtf(var);
        float mx   = maxf[gb];
        float mn   = minf[gb];
        float s    = log10f(dg + 2.0f) / delta;
        float inv  = 1.0f / s;
        float ag[4] = { mean, mx, mn, stdv };
        _Float16* rowp = &At[r * APITCH + d];
        #pragma unroll
        for (int a = 0; a < 4; ++a) {
            rowp[a * 192 + 0]   = (_Float16)(ag[a]);
            rowp[a * 192 + 64]  = (_Float16)(ag[a] * s);
            rowp[a * 192 + 128] = (_Float16)(ag[a] * inv);
        }
    }
    __syncthreads();

    // ---- WMMA main loop
    v8f acc0 = {}, acc1 = {}, acc2 = {}, acc3 = {};
    const int hl   = (lane >> 4) << 3;                 // unused (0 or 8); A loads use koff like B
    const int koff = (lane >> 4) << 4;                 // B: 0 or 16 half offset
    const _Float16* arow = &At[(lane & 15) * APITCH];
    const _Float16* wbase = Wh + (long)(lane & 15) * PNA_K + koff;

    #pragma unroll 4
    for (int t = 0; t < PNA_K / 32; ++t) {
        int kb = t * 32;
        v8h lo = *(const v8h*)(arow + kb + koff);
        v8h hi = *(const v8h*)(arow + kb + koff + 8);
        v16h afrag;
        #pragma unroll
        for (int i = 0; i < 8; ++i) { afrag[i] = lo[i]; afrag[i + 8] = hi[i]; }

        const _Float16* wb = wbase + kb;
        v16h b0 = *(const v16h*)(wb);
        v16h b1 = *(const v16h*)(wb + 16 * PNA_K);
        v16h b2 = *(const v16h*)(wb + 32 * PNA_K);
        v16h b3 = *(const v16h*)(wb + 48 * PNA_K);

        acc0 = __builtin_amdgcn_wmma_f32_16x16x32_f16(false, afrag, false, b0, (short)0, acc0, false, false);
        acc1 = __builtin_amdgcn_wmma_f32_16x16x32_f16(false, afrag, false, b1, (short)0, acc1, false, false);
        acc2 = __builtin_amdgcn_wmma_f32_16x16x32_f16(false, afrag, false, b2, (short)0, acc2, false, false);
        acc3 = __builtin_amdgcn_wmma_f32_16x16x32_f16(false, afrag, false, b3, (short)0, acc3, false, false);
    }

    // ---- bias + leaky relu + store (D layout: lane<16 -> M=v, lane>=16 -> M=v+8; N=lane%16)
    const int n0 = lane & 15;
    const int rb2 = rowbase + ((lane >> 4) << 3);
    const float bs0 = bias[n0], bs1 = bias[16 + n0], bs2 = bias[32 + n0], bs3 = bias[48 + n0];
    #pragma unroll
    for (int vv = 0; vv < 8; ++vv) {
        int mrow = rb2 + vv;
        if (mrow < m) {
            long ob = (long)mrow * PNA_OUT + n0;
            float o;
            o = acc0[vv] + bs0; out[ob]      = o >= 0.f ? o : NEG_SLOPE * o;
            o = acc1[vv] + bs1; out[ob + 16] = o >= 0.f ? o : NEG_SLOPE * o;
            o = acc2[vv] + bs2; out[ob + 32] = o >= 0.f ? o : NEG_SLOPE * o;
            o = acc3[vv] + bs3; out[ob + 48] = o >= 0.f ? o : NEG_SLOPE * o;
        }
    }
}

// ---------------------------------------------------------------- launch
extern "C" void kernel_launch(void* const* d_in, const int* in_sizes, int n_in,
                              void* d_out, int out_size, void* d_ws, size_t ws_size,
                              hipStream_t stream) {
    const int*   rows = (const int*)d_in[0];
    const int*   cols = (const int*)d_in[1];
    const float* vals = (const float*)d_in[2];
    const float* feat = (const float*)d_in[3];
    const float* W    = (const float*)d_in[4];
    const float* bias = (const float*)d_in[5];
    float*       out  = (float*)d_out;

    const int  nnz = in_sizes[0];
    const int  m   = out_size / PNA_OUT;        // 100000
    const long md  = (long)m * PNA_D;

    // workspace carve-up
    float* ws       = (float*)d_ws;
    float* sum      = ws;
    float* sumsq    = sum + md;
    float* maxf     = sumsq + md;
    float* minf     = maxf + md;
    float* deg      = minf + md;                // m floats
    float* misc     = deg + m;                  // [0..RED_BLOCKS) partials, [128] delta
    float* partials = misc;
    float* delta    = misc + 128;
    uintptr_t p = (uintptr_t)(misc + 130);
    p = (p + 63) & ~(uintptr_t)63;              // 64B align for v16h B loads
    _Float16* Wh = (_Float16*)p;                // 64*768 halves

    // 1) zero/initialize accumulators (re-done every call: graph-replay safe)
    {
        long total = md;
        int blocks = (int)((total + 255) / 256);
        pna_init<<<blocks, 256, 0, stream>>>(sum, sumsq, maxf, minf, deg, misc, md, m);
    }
    // 2) W -> f16
    {
        int n = PNA_OUT * PNA_K;
        pna_wconv<<<(n + 255) / 256, 256, 0, stream>>>(W, Wh, n);
    }
    // 3) edge scatter-reduce (one wave per edge)
    {
        long total = (long)nnz * 32;
        int blocks = (int)((total + 255) / 256);
        pna_edges<<<blocks, 256, 0, stream>>>(rows, cols, vals, feat,
                                              sum, sumsq, maxf, minf, deg, nnz);
    }
    // 4) deterministic delta reduction
    pna_logdeg_partial<<<RED_BLOCKS, 256, 0, stream>>>(deg, partials, m);
    pna_logdeg_final<<<1, 32, 0, stream>>>(partials, delta, RED_BLOCKS, m);
    // 5) fused feature-build + WMMA GEMM + bias + leaky relu
    {
        int blocks = (m + 15) / 16;             // 6250
        pna_gemm<<<blocks, 32, 0, stream>>>(sum, sumsq, maxf, minf, deg, delta,
                                            Wh, bias, out, m);
    }
}
